// SSM_12369505812875
// MI455X (gfx1250) — compile-verified, matrix operand order checked by reading, not run on hardware
//
#include <hip/hip_runtime.h>
#include <hip/hip_bf16.h>
#include <math.h>
#include <stdint.h>

typedef __attribute__((ext_vector_type(16))) __bf16 bf16x16;
typedef __attribute__((ext_vector_type(8)))  __bf16 bf16x8;
typedef __attribute__((ext_vector_type(4)))  __bf16 bf16x4;
typedef __attribute__((ext_vector_type(8)))  float  f32x8;
typedef __attribute__((ext_vector_type(2)))  float  f32x2;
typedef __attribute__((ext_vector_type(4)))  unsigned int u32x4;
typedef __attribute__((ext_vector_type(8)))  unsigned int u32x8;

#define BSZ   8
#define L_DIM 2048
#define D_DIM 1024
#define N_DIM 256
#define TCHK  16
#define GCHK  (L_DIM / TCHK)   // 128 chunks
#define MTOT  (BSZ * L_DIM)    // 16384 rows

// B-panel LDS layout: 16B pad after every 1024B of data (matches TDM
// pad_interval=256dw / pad_amount=4dw) -> row stride 1040 bf16 elems (2080B)
#define SB_STRIDE 1040
__device__ __forceinline__ int sb_idx(int r, int k) {
  return r * SB_STRIDE + k + 8 * (k >> 9);
}

// build a 16-elem bf16 WMMA fragment from two contiguous 8-elem (16B) runs
__device__ __forceinline__ bf16x16 frag16(const __bf16* p) {
  bf16x8 a0 = *(const bf16x8*)(p);        // K = kg+0..7
  bf16x8 a1 = *(const bf16x8*)(p + 16);   // K = kg+16..23
  return __builtin_shufflevector(a0, a1, 0, 1, 2, 3, 4, 5, 6, 7,
                                 8, 9, 10, 11, 12, 13, 14, 15);
}

// ---------------------------------------------------------------------------
// Kernel 0: fp32 -> bf16 bulk convert (float4 in, bf16x4 out)
// ---------------------------------------------------------------------------
__global__ __launch_bounds__(256) void k_cvt_bf16(const float* __restrict__ src,
                                                  __bf16* __restrict__ dst, int n4) {
  const int i = blockIdx.x * 256 + threadIdx.x;
  if (i >= n4) return;
  const float4 v = ((const float4*)src)[i];
  bf16x4 o;
  o[0] = (__bf16)v.x; o[1] = (__bf16)v.y; o[2] = (__bf16)v.z; o[3] = (__bf16)v.w;
  ((bf16x4*)dst)[i] = o;
}

// ---------------------------------------------------------------------------
// Kernel 1: uB[m,n] = sum_d x[m,d]*B[n,d]  (bf16 WMMA)
// B panel staged into LDS: rows 0-31 by TDM tensor_load_to_lds (TENSORcnt),
// rows 32-63 by per-lane global_load_async_to_lds_b128 (ASYNCcnt).
// ---------------------------------------------------------------------------
__global__ __launch_bounds__(256) void k_proj_in(const __bf16* __restrict__ xb,
                                                 const __bf16* __restrict__ Bb,
                                                 float* __restrict__ uB) {
  __shared__ __align__(16) __bf16 sB[64 * SB_STRIDE];   // 64-row padded B panel
  const int lane = threadIdx.x & 31, wave = threadIdx.x >> 5;
  const int hi = lane >> 4, lo = lane & 15;
  const int nblk = blockIdx.x & 3;       // 4 N-blocks of 64
  const int mblk = blockIdx.x >> 2;      // 128 M-blocks of 128
  const int m0 = mblk * 128 + wave * 16;
  const int n0base = nblk * 64;
  const int kg = hi * 8;

  // --- TDM: rows 0..31 (32 x 1024 bf16 2D tile, hw-padded 16B per 1024B) ---
  if (threadIdx.x == 0) {
    sB[sb_idx(0, 0) + 1032] = (__bf16)0.f;   // anchor store into pad region
    const unsigned long long ga =
        (unsigned long long)(uintptr_t)(Bb + (size_t)n0base * D_DIM);
    u32x4 g0;
    g0[0] = 1u;                                   // count=1 valid user D#
    g0[1] = (unsigned)(uintptr_t)&sB[0];          // lds_addr
    g0[2] = (unsigned)ga;                         // global_addr[31:0]
    g0[3] = (unsigned)((ga >> 32) & 0x1FFFFFFu) | 0x80000000u;  // addr[56:32]|type=2
    u32x8 g1;
    g1[0] = (1u << 16)    // data_size = 2 bytes
          | (1u << 20)    // pad_enable
          | (7u << 22)    // pad_interval = 256 dwords (1024B)
          | (3u << 25);   // pad_amount   = 4 dwords (16B)
    g1[1] = 1024u << 16;  // tensor_dim0 = 1024
    g1[2] = 32u << 16;    // tensor_dim1 = 32
    g1[3] = 1024u << 16;  // tile_dim0 = 1024
    g1[4] = 32u;          // tile_dim1 = 32 (tile_dim2 = 0)
    g1[5] = 1024u;        // tensor_dim0_stride = 1024
    g1[6] = 0u;
    g1[7] = 0u;
    asm volatile("tensor_load_to_lds %0, %1" :: "s"(g0), "s"(g1) : "memory");
    __builtin_amdgcn_s_wait_tensorcnt(0);
  }

  // --- async per-lane: rows 32..63 (4096 x 16B transfers) ---
  for (int i = threadIdx.x; i < 4096; i += 256) {
    const int row = 32 + (i >> 7), seg = i & 127;
    const __bf16* gp = Bb + (size_t)(n0base + row) * D_DIM + seg * 8;
    const unsigned lds_off = (unsigned)(uintptr_t)&sB[sb_idx(row, seg * 8)];
    asm volatile("global_load_async_to_lds_b128 %0, %1, off"
                 :: "v"(lds_off), "v"((unsigned long long)(uintptr_t)gp)
                 : "memory");
  }
  asm volatile("s_wait_asynccnt 0x0" ::: "memory");
  __syncthreads();

  f32x8 acc[4] = {};
  for (int k0 = 0; k0 < D_DIM; k0 += 32) {
    const __bf16* xrow = xb + (size_t)(m0 + lo) * D_DIM + k0 + kg;
    __builtin_prefetch(xrow + 64, 0, 3);             // global_prefetch_b8 (near)
    const bf16x16 afrag = frag16(xrow);
#pragma unroll
    for (int j = 0; j < 4; ++j) {
      const bf16x16 bfrag = frag16(&sB[sb_idx(j * 16 + lo, k0 + kg)]);
      acc[j] = __builtin_amdgcn_wmma_f32_16x16x32_bf16(false, afrag, false, bfrag,
                                                       (short)0, acc[j], false, false);
    }
  }
#pragma unroll
  for (int j = 0; j < 4; ++j)
#pragma unroll
    for (int i = 0; i < 8; ++i)
      uB[(size_t)(m0 + i + 8 * hi) * N_DIM + n0base + j * 16 + lo] = acc[j][i];
}

// ---------------------------------------------------------------------------
// Kernel 2: W = A^16 via 4 repeated squarings, fp32 WMMA 16x16x4, one workgroup
// ---------------------------------------------------------------------------
__global__ __launch_bounds__(256) void k_apow(const float* __restrict__ A,
                                              float* __restrict__ P0,
                                              float* __restrict__ P1) {
  const int lane = threadIdx.x & 31, wave = threadIdx.x >> 5;
  const int hi = lane >> 4, lo = lane & 15;
  const float* src = A;
  float* dst = P0;
  for (int it = 0; it < 4; ++it) {
    for (int t = wave; t < 256; t += 8) {
      const int mt = t >> 4, nt = t & 15;
      f32x8 acc = {};
      for (int k0 = 0; k0 < N_DIM; k0 += 4) {
        f32x2 a, b;
#pragma unroll
        for (int v = 0; v < 2; ++v) {
          a[v] = src[(size_t)(mt * 16 + lo) * N_DIM + k0 + v + 2 * hi];
          b[v] = src[(size_t)(k0 + v + 2 * hi) * N_DIM + nt * 16 + lo];
        }
        acc = __builtin_amdgcn_wmma_f32_16x16x4_f32(false, a, false, b,
                                                    (short)0, acc, false, false);
      }
#pragma unroll
      for (int i = 0; i < 8; ++i)
        dst[(size_t)(mt * 16 + i + 8 * hi) * N_DIM + nt * 16 + lo] = acc[i];
    }
    __threadfence();
    __syncthreads();
    src = dst;
    dst = (dst == P0) ? P1 : P0;
  }
  // final W = A^16 lands in P1
}

// ---------------------------------------------------------------------------
// Kernel 3: per-chunk local scan (zero init) -> chunk-exit state e_g [8,256]
// ---------------------------------------------------------------------------
__global__ __launch_bounds__(256) void k_chunk_scan(const float* __restrict__ uB,
                                                    const float* __restrict__ A,
                                                    float* __restrict__ e) {
  __shared__ float sLDS[16 * N_DIM];
  const int lane = threadIdx.x & 31, wave = threadIdx.x >> 5;
  const int hi = lane >> 4, lo = lane & 15;
  const int g = blockIdx.x;
  const int col0 = wave * 32;

  for (int i = threadIdx.x; i < 16 * N_DIM; i += 256) sLDS[i] = 0.f;
  __syncthreads();

  for (int t = 0; t < TCHK; ++t) {
    const int l = g * TCHK + t;
    f32x8 acc[2];
#pragma unroll
    for (int j = 0; j < 2; ++j)
#pragma unroll
      for (int i = 0; i < 8; ++i) {
        const int row = i + 8 * hi;
        const int col = col0 + j * 16 + lo;
        acc[j][i] = (row < BSZ) ? uB[((size_t)row * L_DIM + l) * N_DIM + col] : 0.f;
      }
    for (int k0 = 0; k0 < N_DIM; k0 += 4) {
      f32x2 a;
#pragma unroll
      for (int v = 0; v < 2; ++v) a[v] = sLDS[lo * N_DIM + k0 + v + 2 * hi];
#pragma unroll
      for (int j = 0; j < 2; ++j) {
        const int colc = col0 + j * 16 + lo;
        f32x2 b;   // B(k,n) = A^T[k][n] = A[n][k]
#pragma unroll
        for (int v = 0; v < 2; ++v) b[v] = A[(size_t)colc * N_DIM + k0 + v + 2 * hi];
        acc[j] = __builtin_amdgcn_wmma_f32_16x16x4_f32(false, a, false, b,
                                                       (short)0, acc[j], false, false);
      }
    }
    __syncthreads();
#pragma unroll
    for (int j = 0; j < 2; ++j)
#pragma unroll
      for (int i = 0; i < 8; ++i)
        sLDS[(i + 8 * hi) * N_DIM + col0 + j * 16 + lo] = acc[j][i];
    __syncthreads();
  }
  for (int i = threadIdx.x; i < BSZ * N_DIM; i += 256)
    e[(size_t)g * BSZ * N_DIM + i] = sLDS[i];
}

// ---------------------------------------------------------------------------
// Kernel 4: sequential carry chain: c_g = c_{g-1}*(A^16)^T + e_g  (128 steps)
// ---------------------------------------------------------------------------
__global__ __launch_bounds__(256) void k_carry(const float* __restrict__ e,
                                               const float* __restrict__ W,
                                               float* __restrict__ carry) {
  __shared__ float sLDS[16 * N_DIM];
  const int lane = threadIdx.x & 31, wave = threadIdx.x >> 5;
  const int hi = lane >> 4, lo = lane & 15;
  const int col0 = wave * 32;

  for (int i = threadIdx.x; i < 16 * N_DIM; i += 256) sLDS[i] = 0.f;
  __syncthreads();

  for (int g = 0; g < GCHK; ++g) {
    f32x8 acc[2];
#pragma unroll
    for (int j = 0; j < 2; ++j)
#pragma unroll
      for (int i = 0; i < 8; ++i) {
        const int row = i + 8 * hi;
        const int col = col0 + j * 16 + lo;
        acc[j][i] = (row < BSZ) ? e[((size_t)g * BSZ + row) * N_DIM + col] : 0.f;
      }
    for (int k0 = 0; k0 < N_DIM; k0 += 4) {
      f32x2 a;
#pragma unroll
      for (int v = 0; v < 2; ++v) a[v] = sLDS[lo * N_DIM + k0 + v + 2 * hi];
#pragma unroll
      for (int j = 0; j < 2; ++j) {
        const int colc = col0 + j * 16 + lo;
        f32x2 b;   // (A^16)^T[k][n] = W[n][k]
#pragma unroll
        for (int v = 0; v < 2; ++v) b[v] = W[(size_t)colc * N_DIM + k0 + v + 2 * hi];
        acc[j] = __builtin_amdgcn_wmma_f32_16x16x4_f32(false, a, false, b,
                                                       (short)0, acc[j], false, false);
      }
    }
    __syncthreads();
#pragma unroll
    for (int j = 0; j < 2; ++j)
#pragma unroll
      for (int i = 0; i < 8; ++i) {
        const int row = i + 8 * hi;
        const int col = col0 + j * 16 + lo;
        sLDS[row * N_DIM + col] = acc[j][i];
        if (row < BSZ) carry[((size_t)g * BSZ + row) * N_DIM + col] = acc[j][i];
      }
    __syncthreads();
  }
}

// ---------------------------------------------------------------------------
// Kernel 5: re-run chunk scan seeded with carry, stash bf16 states in LDS,
// fused projection y = states * C^T  ([128,1024] tile per chunk) -> d_out
// ---------------------------------------------------------------------------
__global__ __launch_bounds__(256) void k_final(const float* __restrict__ uB,
                                               const float* __restrict__ A,
                                               const __bf16* __restrict__ Cb,
                                               const float* __restrict__ carry,
                                               float* __restrict__ out) {
  __shared__ float  sLDS[16 * N_DIM];
  __shared__ __align__(16) __bf16 st[128][N_DIM + 8];  // 16 steps x 8 rows, padded
  const int lane = threadIdx.x & 31, wave = threadIdx.x >> 5;
  const int hi = lane >> 4, lo = lane & 15;
  const int g = blockIdx.x;
  const int col0 = wave * 32;

  for (int i = threadIdx.x; i < 16 * N_DIM; i += 256) {
    const int row = i >> 8, col = i & (N_DIM - 1);
    sLDS[i] = (g > 0 && row < BSZ) ? carry[((size_t)(g - 1) * BSZ + row) * N_DIM + col] : 0.f;
  }
  __syncthreads();

  // --- scan (fp32 WMMA), capturing states as bf16 ---
  for (int t = 0; t < TCHK; ++t) {
    const int l = g * TCHK + t;
    f32x8 acc[2];
#pragma unroll
    for (int j = 0; j < 2; ++j)
#pragma unroll
      for (int i = 0; i < 8; ++i) {
        const int row = i + 8 * hi;
        const int col = col0 + j * 16 + lo;
        acc[j][i] = (row < BSZ) ? uB[((size_t)row * L_DIM + l) * N_DIM + col] : 0.f;
      }
    for (int k0 = 0; k0 < N_DIM; k0 += 4) {
      f32x2 a;
#pragma unroll
      for (int v = 0; v < 2; ++v) a[v] = sLDS[lo * N_DIM + k0 + v + 2 * hi];
#pragma unroll
      for (int j = 0; j < 2; ++j) {
        const int colc = col0 + j * 16 + lo;
        f32x2 b;
#pragma unroll
        for (int v = 0; v < 2; ++v) b[v] = A[(size_t)colc * N_DIM + k0 + v + 2 * hi];
        acc[j] = __builtin_amdgcn_wmma_f32_16x16x4_f32(false, a, false, b,
                                                       (short)0, acc[j], false, false);
      }
    }
    __syncthreads();
#pragma unroll
    for (int j = 0; j < 2; ++j)
#pragma unroll
      for (int i = 0; i < 8; ++i) {
        const int row = i + 8 * hi;
        const int col = col0 + j * 16 + lo;
        sLDS[row * N_DIM + col] = acc[j][i];
        if (row < BSZ) st[t * BSZ + row][col] = (__bf16)acc[j][i];
      }
    __syncthreads();
  }

  // --- projection: [128,256] x [256,1024] bf16 WMMA; wave w owns M-tile w ---
  const int mrow = wave * 16 + lo;
  const int kg = hi * 8;
  for (int nt = 0; nt < 64; ++nt) {
    const int n0 = nt * 16;
    f32x8 acc = {};
    for (int k0 = 0; k0 < N_DIM; k0 += 32) {
      const bf16x16 afrag = frag16(&st[mrow][k0 + kg]);
      const bf16x16 bfrag = frag16(Cb + (size_t)(n0 + lo) * N_DIM + k0 + kg);
      acc = __builtin_amdgcn_wmma_f32_16x16x32_bf16(false, afrag, false, bfrag,
                                                    (short)0, acc, false, false);
    }
#pragma unroll
    for (int i = 0; i < 8; ++i) {
      const int r = wave * 16 + i + 8 * hi;   // 0..127 = t*8+b
      const int t = r >> 3, b = r & 7;
      const int l = g * TCHK + t;
      out[((size_t)b * L_DIM + l) * D_DIM + n0 + lo] = acc[i];
    }
  }
}

// ---------------------------------------------------------------------------
// Kernel 6: in-place exact-erf GELU + LayerNorm + nan_to_num, one block per row
// ---------------------------------------------------------------------------
__global__ __launch_bounds__(256) void k_ln(float* __restrict__ out,
                                            const float* __restrict__ gamma,
                                            const float* __restrict__ beta) {
  __shared__ float red[16];
  float* p = out + (size_t)blockIdx.x * D_DIM;
  float a[4], s = 0.f, s2 = 0.f;
#pragma unroll
  for (int j = 0; j < 4; ++j) {
    const int d = threadIdx.x + j * 256;
    const float v = p[d];
    const float ge = 0.5f * v * (1.0f + erff(v * 0.70710678118654752f));
    a[j] = ge; s += ge; s2 += ge * ge;
  }
#pragma unroll
  for (int off = 16; off; off >>= 1) {       // wave32 reduce
    s  += __shfl_xor(s, off, 32);
    s2 += __shfl_xor(s2, off, 32);
  }
  const int wave = threadIdx.x >> 5;
  if ((threadIdx.x & 31) == 0) { red[wave] = s; red[8 + wave] = s2; }
  __syncthreads();
  if (threadIdx.x < 32) {
    float t  = (threadIdx.x < 8) ? red[threadIdx.x] : 0.f;
    float t2 = (threadIdx.x < 8) ? red[8 + threadIdx.x] : 0.f;
#pragma unroll
    for (int off = 4; off; off >>= 1) {
      t += __shfl_xor(t, off, 32);
      t2 += __shfl_xor(t2, off, 32);
    }
    if (threadIdx.x == 0) { red[0] = t; red[1] = t2; }
  }
  __syncthreads();
  const float mean = red[0] * (1.0f / D_DIM);
  const float var  = red[1] * (1.0f / D_DIM) - mean * mean;
  const float inv  = rsqrtf(var + 1e-5f);
#pragma unroll
  for (int j = 0; j < 4; ++j) {
    const int d = threadIdx.x + j * 256;
    float v = (a[j] - mean) * inv * gamma[d] + beta[d];
    if (__builtin_isnan(v)) v = 0.f;
    v = fminf(fmaxf(v, -1e6f), 1e6f);
    p[d] = v;
  }
}

// ---------------------------------------------------------------------------
extern "C" void kernel_launch(void* const* d_in, const int* in_sizes, int n_in,
                              void* d_out, int out_size, void* d_ws, size_t ws_size,
                              hipStream_t stream) {
  (void)in_sizes; (void)n_in; (void)out_size; (void)ws_size;
  const float* x     = (const float*)d_in[0];
  const float* A     = (const float*)d_in[1];
  const float* Bm    = (const float*)d_in[2];
  const float* Cm    = (const float*)d_in[3];
  const float* gamma = (const float*)d_in[4];
  const float* beta  = (const float*)d_in[5];
  float* out = (float*)d_out;

  char* ws = (char*)d_ws;
  float*  uB    = (float*) (ws);                      // 16 MB
  float*  P0    = (float*) (ws + 16777216);           // 256 KB
  float*  P1    = (float*) (ws + 17039360);           // 256 KB (W = A^16)
  float*  e     = (float*) (ws + 17301504);           // 1 MB
  float*  carry = (float*) (ws + 18350080);           // 1 MB
  __bf16* xb    = (__bf16*)(ws + 19398656);           // 32 MB
  __bf16* Bb    = (__bf16*)(ws + 52953088);           // 512 KB
  __bf16* Cb    = (__bf16*)(ws + 53477376);           // 512 KB

  k_cvt_bf16<<<16384, 256, 0, stream>>>(x,  xb, (MTOT * D_DIM) / 4);
  k_cvt_bf16<<<256,   256, 0, stream>>>(Bm, Bb, (N_DIM * D_DIM) / 4);
  k_cvt_bf16<<<256,   256, 0, stream>>>(Cm, Cb, (D_DIM * N_DIM) / 4);

  k_proj_in<<<512, 256, 0, stream>>>(xb, Bb, uB);
  k_apow<<<1, 256, 0, stream>>>(A, P0, P1);
  k_chunk_scan<<<GCHK, 256, 0, stream>>>(uB, A, e);
  k_carry<<<1, 256, 0, stream>>>(e, P1, carry);
  k_final<<<GCHK, 256, 0, stream>>>(uB, A, Cb, carry, out);
  k_ln<<<MTOT, 256, 0, stream>>>(out, gamma, beta);
}
